// TKMDAttention_4793183502633
// MI455X (gfx1250) — compile-verified, matrix operand order checked by reading, not run on hardware
//
#include <hip/hip_runtime.h>
#include <hip/hip_bf16.h>

typedef __attribute__((ext_vector_type(16))) __bf16 v16bf;
typedef __attribute__((ext_vector_type(8)))  __bf16 bf16x8;
typedef __attribute__((ext_vector_type(8)))  float  v8f;
typedef __attribute__((ext_vector_type(4)))  float  f32x4;

#define DIM   384
#define HEADS 8
#define CH    48
#define HW    16384
#define BATCH 2
#define CSPLIT 64          // split-K chunks for channel attention (16384/64 = 256)

// 16-bit WMMA operand: lane holds A[row][half*8 + 0..7] in elems 0..7 and
// A[row][16 + half*8 + 0..7] in elems 8..15  (row = lane&15, half = lane>>4)
__device__ __forceinline__ v16bf frag_cat(bf16x8 lo, bf16x8 hi) {
  return __builtin_shufflevector(lo, hi, 0, 1, 2, 3, 4, 5, 6, 7,
                                 8, 9, 10, 11, 12, 13, 14, 15);
}

// gfx1250 async global->LDS copy (ASYNCcnt-tracked), GVS addressing
__device__ __forceinline__ void async_load_b128(unsigned lds_off, const void* sbase,
                                                int voff) {
  asm volatile("global_load_async_to_lds_b128 %0, %1, %2"
               :: "v"(lds_off), "v"(voff), "s"(sbase) : "memory");
}
__device__ __forceinline__ void wait_async() {
  asm volatile("s_wait_asynccnt 0x0" ::: "memory");
}

// ---------------------------------------------------------------- converts
__global__ void cvt_f32_bf16(const float* __restrict__ in, __bf16* __restrict__ out, int n) {
  int i = blockIdx.x * blockDim.x + threadIdx.x;
  if (i < n) out[i] = (__bf16)in[i];
}

// tiled transpose + convert: in [rows][cols] f32 -> outT [cols][rows] bf16
__global__ void transpose_cvt_bf16(const float* __restrict__ in, __bf16* __restrict__ outT,
                                   int rows, int cols) {
  __shared__ __bf16 T[64][65];
  int cb = blockIdx.y * 64;            // row (channel) base
  int pb = blockIdx.x * 64;            // col (pixel) base
  const float* ip = in + (long)blockIdx.z * rows * cols;
  __bf16* op = outT + (long)blockIdx.z * rows * cols;
  int tid = threadIdx.x;
  for (int idx = tid; idx < 64 * 64; idx += 256) {
    int c = idx >> 6, pp = idx & 63;   // consecutive tid -> consecutive pixel
    T[c][pp] = (__bf16)ip[(long)(cb + c) * cols + pb + pp];
  }
  __syncthreads();
  for (int idx = tid; idx < 64 * 8; idx += 256) {
    int pp = idx >> 3, ck = (idx & 7) * 8;
    bf16x8 vv;
#pragma unroll
    for (int e = 0; e < 8; ++e) vv[e] = T[ck + e][pp];
    *(bf16x8*)&op[(long)(pb + pp) * rows + cb + ck] = vv;
  }
}

// --------- big GEMM: C = A(MxK) * BT(NxK)^T, all staging via async->LDS
// block = 256 threads = 8 waves (2M x 4N), each wave 2x2 WMMA tiles (32x32)
__global__ void gemm_bf16_wmma(const __bf16* __restrict__ A, const __bf16* __restrict__ BT,
                               float* __restrict__ C, int M, int N, int K,
                               long strideB, long strideC) {
  __shared__ __bf16 As[64][32];     // M x K slab
  __shared__ __bf16 Bs[128][32];    // N x K slab
  const int tid = threadIdx.x;
  const int lane = tid & 31, wave = tid >> 5;
  const int wm = wave >> 2, wn = wave & 3;
  const int m0 = blockIdx.y * 64;
  const int n0 = blockIdx.x * 128;
  const __bf16* Bb = BT + (long)blockIdx.z * strideB;
  float* Cb = C + (long)blockIdx.z * strideC;
  const int rc = lane & 15, half = lane >> 4;
  const int arow = tid >> 2, acol = (tid & 3) * 8;   // A: 1 x b128 per thread
  const int brow = tid >> 1, bcol = (tid & 1) * 16;  // B: 2 x b128 per thread
  v8f acc[2][2] = {};
  for (int k0 = 0; k0 < K; k0 += 32) {
    if (k0) __syncthreads();                         // protect LDS reuse
    async_load_b128((unsigned)(size_t)&As[arow][acol], (const void*)A,
                    (int)(((long)(m0 + arow) * K + k0 + acol) * 2));
    async_load_b128((unsigned)(size_t)&Bs[brow][bcol], (const void*)Bb,
                    (int)(((long)(n0 + brow) * K + k0 + bcol) * 2));
    async_load_b128((unsigned)(size_t)&Bs[brow][bcol + 8], (const void*)Bb,
                    (int)(((long)(n0 + brow) * K + k0 + bcol + 8) * 2));
    wait_async();
    __syncthreads();
    v16bf a[2], b[2];
#pragma unroll
    for (int i = 0; i < 2; ++i) {
      a[i] = frag_cat(*(const bf16x8*)&As[wm * 32 + i * 16 + rc][half * 8],
                      *(const bf16x8*)&As[wm * 32 + i * 16 + rc][16 + half * 8]);
      b[i] = frag_cat(*(const bf16x8*)&Bs[wn * 32 + i * 16 + rc][half * 8],
                      *(const bf16x8*)&Bs[wn * 32 + i * 16 + rc][16 + half * 8]);
    }
#pragma unroll
    for (int i = 0; i < 2; ++i)
#pragma unroll
      for (int j = 0; j < 2; ++j)
        acc[i][j] = __builtin_amdgcn_wmma_f32_16x16x32_bf16(false, a[i], false, b[j],
                                                            (short)0, acc[i][j],
                                                            false, false);
  }
#pragma unroll
  for (int i = 0; i < 2; ++i)
#pragma unroll
    for (int j = 0; j < 2; ++j)
#pragma unroll
      for (int r = 0; r < 8; ++r)
        Cb[(long)(m0 + wm * 32 + i * 16 + r + 8 * half) * N +
           n0 + wn * 32 + j * 16 + rc] = acc[i][j][r];
}

// ---------------------------------------------------------- depthwise 3x3
__global__ void dwconv3x3(const float* __restrict__ in, const float* __restrict__ w,
                          float* __restrict__ out, int total) {
  int i = blockIdx.x * blockDim.x + threadIdx.x;
  if (i >= total) return;
  int x = i & 127, y = (i >> 7) & 127;
  int plane = i >> 14;               // b*1152 + c
  int c = plane % (3 * DIM);
  const float* wp = w + c * 9;
  const float* ip = in + ((long)plane << 14);
  float s = 0.f;
#pragma unroll
  for (int ky = 0; ky < 3; ++ky) {
    int yy = y + ky - 1;
    if (yy < 0 || yy > 127) continue;
#pragma unroll
    for (int kx = 0; kx < 3; ++kx) {
      int xx = x + kx - 1;
      if (xx < 0 || xx > 127) continue;
      s += ip[yy * 128 + xx] * wp[ky * 3 + kx];
    }
  }
  out[i] = s;
}

// --------------------------------------- per-channel inverse L2 norm (q,k)
__global__ void l2norm_rows(const float* __restrict__ qkv, float* __restrict__ invq,
                            float* __restrict__ invk) {
  int id = blockIdx.x;                 // BATCH*2*DIM rows
  int ch = id % DIM;
  int sel = (id / DIM) & 1;            // 0=q, 1=k
  int b = id / (2 * DIM);
  const float* p = qkv + ((long)(b * 3 * DIM + sel * DIM + ch) << 14);
  __shared__ float red[256];
  float s = 0.f;
  for (int i = threadIdx.x; i < HW; i += 256) { float v = p[i]; s += v * v; }
  red[threadIdx.x] = s;
  __syncthreads();
  for (int st = 128; st; st >>= 1) {
    if (threadIdx.x < st) red[threadIdx.x] += red[threadIdx.x + st];
    __syncthreads();
  }
  if (threadIdx.x == 0)
    (sel ? invk : invq)[b * DIM + ch] = 1.f / fmaxf(sqrtf(red[0]), 1e-12f);
}

// --------- channel attention partials: split-K over 16384 (deterministic)
// grid = 16*CSPLIT, block = 288 (9 waves -> 3x3 tiles of 16x16)
__global__ void chan_attn_part(const float* __restrict__ qkv, const float* __restrict__ invq,
                               const float* __restrict__ invk,
                               float* __restrict__ part) {
  int blk = blockIdx.x;
  int bh = blk & 15, chunk = blk >> 4;
  int head = bh & 7, b = bh >> 3;
  int lane = threadIdx.x & 31, wave = threadIdx.x >> 5;
  const float* qp = qkv + (((long)(b * 3 * DIM) + head * CH) << 14);
  const float* kp = qkv + (((long)(b * 3 * DIM + DIM) + head * CH) << 14);
  const float* iq = invq + b * DIM + head * CH;
  const float* ik = invk + b * DIM + head * CH;
  int rc = lane & 15, half = lane >> 4;
  int tm = wave / 3, tn = wave % 3;
  float sA = iq[tm * 16 + rc];
  float sB = ik[tn * 16 + rc];
  const float* qrow = qp + (long)(tm * 16 + rc) * HW;
  const float* krow = kp + (long)(tn * 16 + rc) * HW;
  const int kbeg = chunk * (HW / CSPLIT), kend = kbeg + HW / CSPLIT;
  v8f acc = {};
  for (int k0 = kbeg; k0 < kend; k0 += 32) {
    f32x4 qa = *(const f32x4*)&qrow[k0 + half * 8];
    f32x4 qb = *(const f32x4*)&qrow[k0 + half * 8 + 4];
    f32x4 qc = *(const f32x4*)&qrow[k0 + 16 + half * 8];
    f32x4 qd = *(const f32x4*)&qrow[k0 + 16 + half * 8 + 4];
    f32x4 ka = *(const f32x4*)&krow[k0 + half * 8];
    f32x4 kb = *(const f32x4*)&krow[k0 + half * 8 + 4];
    f32x4 kc = *(const f32x4*)&krow[k0 + 16 + half * 8];
    f32x4 kd = *(const f32x4*)&krow[k0 + 16 + half * 8 + 4];
    v16bf a, bb;
#pragma unroll
    for (int e = 0; e < 4; ++e) {
      a[e]       = (__bf16)(qa[e] * sA);
      a[4 + e]   = (__bf16)(qb[e] * sA);
      a[8 + e]   = (__bf16)(qc[e] * sA);
      a[12 + e]  = (__bf16)(qd[e] * sA);
      bb[e]      = (__bf16)(ka[e] * sB);
      bb[4 + e]  = (__bf16)(kb[e] * sB);
      bb[8 + e]  = (__bf16)(kc[e] * sB);
      bb[12 + e] = (__bf16)(kd[e] * sB);
    }
    acc = __builtin_amdgcn_wmma_f32_16x16x32_bf16(false, a, false, bb, (short)0, acc,
                                                  false, false);
  }
  float* pp = part + (long)(chunk * 16 + bh) * (48 * 48);
#pragma unroll
  for (int r = 0; r < 8; ++r)
    pp[(tm * 16 + r + 8 * half) * 48 + tn * 16 + rc] = acc[r];
}

// combine split-K partials, then per-row topk {24,32,36,38} softmax blend
__global__ void chan_combine(const float* __restrict__ part, const float* __restrict__ temp,
                             const float* __restrict__ a1, const float* __restrict__ a2,
                             const float* __restrict__ a3, const float* __restrict__ a4,
                             float* __restrict__ Wc) {
  int bh = blockIdx.x, head = bh & 7;
  __shared__ float catt[48][48];
  __shared__ float scr[48][48];
  int tid = threadIdx.x;
  float T = temp[head];
  for (int idx = tid; idx < 48 * 48; idx += 256) {
    float s = 0.f;
    for (int c = 0; c < CSPLIT; ++c)
      s += part[(long)(c * 16 + bh) * (48 * 48) + idx];
    catt[idx / 48][idx % 48] = s * T;
  }
  __syncthreads();
  if (tid < 48) {
    float* rv = scr[tid];
    for (int j = 0; j < 48; ++j) rv[j] = catt[tid][j];
    for (int i = 0; i < 38; ++i) {              // partial selection sort (max k=38)
      int mx = i;
      for (int j = i + 1; j < 48; ++j) if (rv[j] > rv[mx]) mx = j;
      float t = rv[i]; rv[i] = rv[mx]; rv[mx] = t;
    }
    float thr0 = rv[23], thr1 = rv[31], thr2 = rv[35], thr3 = rv[37];
    float m = rv[0];
    float w0 = a1[0], w1 = a2[0], w2 = a3[0], w3 = a4[0];
    float S0 = 0, S1 = 0, S2 = 0, S3 = 0;
    for (int j = 0; j < 48; ++j) {
      float av = catt[tid][j], e = __expf(av - m);
      if (av >= thr0) S0 += e;
      if (av >= thr1) S1 += e;
      if (av >= thr2) S2 += e;
      if (av >= thr3) S3 += e;
    }
    float i0 = w0 / S0, i1 = w1 / S1, i2 = w2 / S2, i3 = w3 / S3;
    for (int j = 0; j < 48; ++j) {
      float av = catt[tid][j], e = __expf(av - m), wv = 0.f;
      if (av >= thr0) wv += i0 * e;
      if (av >= thr1) wv += i1 * e;
      if (av >= thr2) wv += i2 * e;
      if (av >= thr3) wv += i3 * e;
      Wc[((long)bh * 48 + tid) * 48 + j] = wv;
    }
  }
}

// ---------------- cout = Wc(48x48) @ v(48x16384): LDS-staged, 48x128 per block
// grid = (HW/128, 16 b*head), block = 256 (8 waves -> 8 n-subtiles)
__global__ void chan_out(const float* __restrict__ Wc, const float* __restrict__ qkv,
                         float* __restrict__ cout) {
  __shared__ __bf16 vTs[128][64];   // v transposed [pix][ch], K-padded
  __shared__ __bf16 Wcs[48][64];    // Wc rows, K-padded
  int bh = blockIdx.y, head = bh & 7, b = bh >> 3;
  int n0 = blockIdx.x * 128;
  int tid = threadIdx.x;
  const float* vp = qkv + (((long)(b * 3 * DIM + 2 * DIM) + head * CH) << 14);
  const float* Wp = Wc + (long)bh * 48 * 48;
  for (int idx = tid; idx < 48 * 128; idx += 256) {
    int c = idx >> 7, pp = idx & 127;     // coalesced global read
    vTs[pp][c] = (__bf16)vp[((long)c << 14) + n0 + pp];
  }
  for (int idx = tid; idx < 128 * 16; idx += 256)
    vTs[idx >> 4][48 + (idx & 15)] = (__bf16)0.f;
  for (int idx = tid; idx < 48 * 64; idx += 256) {
    int rr = idx >> 6, cc = idx & 63;
    Wcs[rr][cc] = (cc < 48) ? (__bf16)Wp[rr * 48 + cc] : (__bf16)0.f;
  }
  __syncthreads();
  int lane = tid & 31, wave = tid >> 5;
  int rc = lane & 15, half = lane >> 4;
  for (int tm = 0; tm < 3; ++tm) {
    v8f acc = {};
#pragma unroll
    for (int k0 = 0; k0 < 64; k0 += 32) {
      v16bf a = frag_cat(*(const bf16x8*)&Wcs[tm * 16 + rc][k0 + half * 8],
                         *(const bf16x8*)&Wcs[tm * 16 + rc][k0 + 16 + half * 8]);
      v16bf bb = frag_cat(*(const bf16x8*)&vTs[wave * 16 + rc][k0 + half * 8],
                          *(const bf16x8*)&vTs[wave * 16 + rc][k0 + 16 + half * 8]);
      acc = __builtin_amdgcn_wmma_f32_16x16x32_bf16(false, a, false, bb, (short)0, acc,
                                                    false, false);
    }
#pragma unroll
    for (int r = 0; r < 8; ++r) {
      int cg = head * CH + tm * 16 + r + 8 * half;
      cout[(((long)b * DIM + cg) << 14) + n0 + wave * 16 + rc] = acc[r];
    }
  }
}

// ---------------- spatial window attention, fully fused per (b, window, head)
// grid = 4096, block = 256 (8 waves); writes mixT [b][pix][ch] (bf16)
__global__ void spat_attn(const float* __restrict__ qkv, const float* __restrict__ cout,
                          const float* __restrict__ temp,
                          const float* __restrict__ a1, const float* __restrict__ a2,
                          const float* __restrict__ a3, const float* __restrict__ a4,
                          __bf16* __restrict__ mixT) {
  __shared__ __bf16 Aq[64][64];     // q normalized [token][ch], K-padded
  __shared__ __bf16 Bk[64][64];     // k normalized [token][ch], K-padded
  __shared__ __bf16 Vt[48][64];     // v transposed [ch][token]
  __shared__ __bf16 Wsb[64][64];    // combined attention weights (bf16)
  __shared__ float Satt[64][64];
  __shared__ float Scr[64][64];
  __shared__ float invqs[64], invks[64];
  int u = blockIdx.x;
  int head = u & 7, win = (u >> 3) & 255, b = u >> 11;
  int hb = win >> 4, wb = win & 15;
  const long qoff = ((long)(b * 3 * DIM) + head * CH) << 14;
  const long koff = qoff + ((long)DIM << 14);
  const long voff = qoff + ((long)(2 * DIM) << 14);
  int tid = threadIdx.x;
  if (tid < 64) {                                   // token norms over 48 ch
    int dy = tid >> 3, dx = tid & 7;
    long pix = (long)(hb * 8 + dy) * 128 + wb * 8 + dx;
    float sq = 0, sk = 0;
    for (int c = 0; c < 48; ++c) {
      float qv = qkv[qoff + ((long)c << 14) + pix]; sq += qv * qv;
      float kv = qkv[koff + ((long)c << 14) + pix]; sk += kv * kv;
    }
    invqs[tid] = 1.f / fmaxf(sqrtf(sq), 1e-12f);
    invks[tid] = 1.f / fmaxf(sqrtf(sk), 1e-12f);
  }
  __syncthreads();
  for (int idx = tid; idx < 64 * 48; idx += 256) {  // stage tiles to LDS (bf16)
    int t = idx / 48, c = idx % 48;
    int dy = t >> 3, dx = t & 7;
    long pix = (long)(hb * 8 + dy) * 128 + wb * 8 + dx;
    Aq[t][c] = (__bf16)(qkv[qoff + ((long)c << 14) + pix] * invqs[t]);
    Bk[t][c] = (__bf16)(qkv[koff + ((long)c << 14) + pix] * invks[t]);
    Vt[c][t] = (__bf16)qkv[voff + ((long)c << 14) + pix];
  }
  for (int idx = tid; idx < 64 * 16; idx += 256) {  // zero K padding
    int t = idx >> 4, c = 48 + (idx & 15);
    Aq[t][c] = (__bf16)0.f;
    Bk[t][c] = (__bf16)0.f;
  }
  __syncthreads();
  int lane = tid & 31, wave = tid >> 5;
  int rc = lane & 15, half = lane >> 4;
  float T = temp[head];
  for (int tt = wave; tt < 16; tt += 8) {           // sattn = qn @ kn^T
    int tm = tt >> 2, tn = tt & 3;
    v8f acc = {};
#pragma unroll
    for (int k0 = 0; k0 < 64; k0 += 32) {
      v16bf a = frag_cat(*(const bf16x8*)&Aq[tm * 16 + rc][k0 + half * 8],
                         *(const bf16x8*)&Aq[tm * 16 + rc][k0 + 16 + half * 8]);
      v16bf bb = frag_cat(*(const bf16x8*)&Bk[tn * 16 + rc][k0 + half * 8],
                          *(const bf16x8*)&Bk[tn * 16 + rc][k0 + 16 + half * 8]);
      acc = __builtin_amdgcn_wmma_f32_16x16x32_bf16(false, a, false, bb, (short)0, acc,
                                                    false, false);
    }
#pragma unroll
    for (int r = 0; r < 8; ++r)
      Satt[tm * 16 + r + 8 * half][tn * 16 + rc] = acc[r] * T;
  }
  __syncthreads();
  if (tid < 64) {                                   // topk {32,42,48,51} combine
    float* rv = Scr[tid];
    for (int j = 0; j < 64; ++j) rv[j] = Satt[tid][j];
    for (int i = 0; i < 51; ++i) {
      int mx = i;
      for (int j = i + 1; j < 64; ++j) if (rv[j] > rv[mx]) mx = j;
      float t = rv[i]; rv[i] = rv[mx]; rv[mx] = t;
    }
    float thr0 = rv[31], thr1 = rv[41], thr2 = rv[47], thr3 = rv[50];
    float m = rv[0];
    float w0 = a1[0], w1 = a2[0], w2 = a3[0], w3 = a4[0];
    float S0 = 0, S1 = 0, S2 = 0, S3 = 0;
    for (int j = 0; j < 64; ++j) {
      float av = Satt[tid][j], e = __expf(av - m);
      if (av >= thr0) S0 += e;
      if (av >= thr1) S1 += e;
      if (av >= thr2) S2 += e;
      if (av >= thr3) S3 += e;
    }
    float i0 = w0 / S0, i1 = w1 / S1, i2 = w2 / S2, i3 = w3 / S3;
    for (int j = 0; j < 64; ++j) {
      float av = Satt[tid][j], e = __expf(av - m), wv = 0.f;
      if (av >= thr0) wv += i0 * e;
      if (av >= thr1) wv += i1 * e;
      if (av >= thr2) wv += i2 * e;
      if (av >= thr3) wv += i3 * e;
      Wsb[tid][j] = (__bf16)wv;
    }
  }
  __syncthreads();
  for (int tt = wave; tt < 12; tt += 8) {           // sout = W @ v, + channel branch
    int tm = tt / 3, tn = tt % 3;
    v8f acc = {};
#pragma unroll
    for (int k0 = 0; k0 < 64; k0 += 32) {
      v16bf a = frag_cat(*(const bf16x8*)&Wsb[tm * 16 + rc][k0 + half * 8],
                         *(const bf16x8*)&Wsb[tm * 16 + rc][k0 + 16 + half * 8]);
      v16bf bb = frag_cat(*(const bf16x8*)&Vt[tn * 16 + rc][k0 + half * 8],
                          *(const bf16x8*)&Vt[tn * 16 + rc][k0 + 16 + half * 8]);
      acc = __builtin_amdgcn_wmma_f32_16x16x32_bf16(false, a, false, bb, (short)0, acc,
                                                    false, false);
    }
#pragma unroll
    for (int r = 0; r < 8; ++r) {
      int t = tm * 16 + r + 8 * half;
      int dy = t >> 3, dx = t & 7;
      long pix = (long)(hb * 8 + dy) * 128 + wb * 8 + dx;
      int cg = head * CH + tn * 16 + rc;
      float cv = cout[(((long)b * DIM + cg) << 14) + pix];
      mixT[((long)b * HW + pix) * DIM + cg] = (__bf16)(acc[r] + cv);
    }
  }
}

// ------------------------------------------------------------- launcher
extern "C" void kernel_launch(void* const* d_in, const int* in_sizes, int n_in,
                              void* d_out, int out_size, void* d_ws, size_t ws_size,
                              hipStream_t stream) {
  const float* x      = (const float*)d_in[0];
  const float* w_qkv  = (const float*)d_in[1];
  const float* w_dw   = (const float*)d_in[2];
  const float* w_proj = (const float*)d_in[3];
  const float* temp   = (const float*)d_in[4];
  const float* a1     = (const float*)d_in[5];
  const float* a2     = (const float*)d_in[6];
  const float* a3     = (const float*)d_in[7];
  const float* a4     = (const float*)d_in[8];
  float* out = (float*)d_out;

  const size_t nX    = (size_t)BATCH * DIM * HW;          // 12.6M
  const size_t nQKV  = (size_t)BATCH * 3 * DIM * HW;      // 37.7M
  char* p = (char*)d_ws;
  size_t o = 0;
  auto take = [&](size_t bytes) { size_t r = o; o = (o + bytes + 255) & ~(size_t)255; return r; };
  __bf16* xT      = (__bf16*)(p + take(nX * 2));                       // [b][pix][ch]
  __bf16* wqkvbf  = (__bf16*)(p + take((size_t)3 * DIM * DIM * 2));
  __bf16* wprojbf = (__bf16*)(p + take((size_t)DIM * DIM * 2));
  float*  qkvpre  = (float*)(p + take(nQKV * 4));
  float*  qkvbuf  = (float*)(p + take(nQKV * 4));
  float*  invq    = (float*)(p + take((size_t)BATCH * DIM * 4));
  float*  invk    = (float*)(p + take((size_t)BATCH * DIM * 4));
  float*  cpart   = (float*)(p + take((size_t)CSPLIT * 16 * 48 * 48 * 4));
  float*  Wc      = (float*)(p + take((size_t)BATCH * HEADS * 48 * 48 * 4));
  float*  coutb   = (float*)(p + take(nX * 4));
  __bf16* mixT    = (__bf16*)(p + take(nX * 2));                       // [b][pix][ch]
  (void)ws_size; (void)n_in; (void)in_sizes; (void)out_size;

  // 1) weight converts + x transpose-convert (GEMM B operands become [N][K])
  cvt_f32_bf16<<<(3 * DIM * DIM + 255) / 256, 256, 0, stream>>>(w_qkv, wqkvbf, 3 * DIM * DIM);
  cvt_f32_bf16<<<(DIM * DIM + 255) / 256, 256, 0, stream>>>(w_proj, wprojbf, DIM * DIM);
  transpose_cvt_bf16<<<dim3(HW / 64, DIM / 64, BATCH), 256, 0, stream>>>(x, xT, DIM, HW);

  // 2) qkv GEMM: (1152x384) x (384x16384) per batch
  gemm_bf16_wmma<<<dim3(HW / 128, (3 * DIM) / 64, BATCH), 256, 0, stream>>>(
      wqkvbf, xT, qkvpre, 3 * DIM, HW, DIM, (long)DIM * HW, (long)3 * DIM * HW);

  // 3) depthwise 3x3
  dwconv3x3<<<(int)((nQKV + 255) / 256), 256, 0, stream>>>(qkvpre, w_dw, qkvbuf, (int)nQKV);

  // 4) channel norms for q,k
  l2norm_rows<<<BATCH * 2 * DIM, 256, 0, stream>>>(qkvbuf, invq, invk);

  // 5) channel attention: split-K partials + combine/topk
  chan_attn_part<<<16 * CSPLIT, 288, 0, stream>>>(qkvbuf, invq, invk, cpart);
  chan_combine<<<16, 256, 0, stream>>>(cpart, temp, a1, a2, a3, a4, Wc);

  // 6) channel branch output
  chan_out<<<dim3(HW / 128, 16), 256, 0, stream>>>(Wc, qkvbuf, coutb);

  // 7) spatial attention (+ add channel branch, emit bf16 proj input, transposed)
  spat_attn<<<BATCH * 256 * HEADS, 256, 0, stream>>>(qkvbuf, coutb, temp, a1, a2, a3, a4, mixT);

  // 8) projection GEMM -> fp32 output
  gemm_bf16_wmma<<<dim3(HW / 128, DIM / 64, BATCH), 256, 0, stream>>>(
      wprojbf, mixT, out, DIM, HW, DIM, (long)DIM * HW, (long)DIM * HW);
}
